// LCGN_BACKBONE_35003983462642
// MI455X (gfx1250) — compile-verified, matrix operand order checked by reading, not run on hardware
//
#include <hip/hip_runtime.h>
#include <hip/hip_bf16.h>
#include <math.h>

#define BB   256
#define NNE  100
#define DFE  2112
#define CTXE 512
#define CMDE 512
#define LLE  30
#define TTE  4
#define ROWS (BB * NNE)     /* 25600 */
#define NEGV (-1e30f)

typedef __attribute__((ext_vector_type(16))) __bf16 bf16x16;
typedef __attribute__((ext_vector_type(8)))  __bf16 bf16x8;
typedef __attribute__((ext_vector_type(8)))  float  f32x8;

// ---------------------------------------------------------------------------
// WMMA fragment loaders (layout per CDNA5 ISA 7.12.2, 16-bit A 16x32)
// ---------------------------------------------------------------------------
__device__ __forceinline__ bf16x16 frag_ld(const __bf16* base, int ld,
                                           int row0, int k0, int lane) {
  int m = lane & 15, h = lane >> 4;
  const __bf16* p = base + (long)(row0 + m) * ld + k0 + 8 * h;
  bf16x8 c0 = *(const bf16x8*)p;
  bf16x8 c1 = *(const bf16x8*)(p + 16);
  bf16x16 f;
#pragma unroll
  for (int i = 0; i < 8; ++i) { f[i] = c0[i]; f[8 + i] = c1[i]; }
  return f;
}

__device__ __forceinline__ bf16x16 frag_ld_clamp(const __bf16* base, int ld,
                                                 int row0, int k0, int lane,
                                                 int maxrow) {
  int m = lane & 15, h = lane >> 4;
  int r = row0 + m; if (r > maxrow - 1) r = maxrow - 1;
  const __bf16* p = base + (long)r * ld + k0 + 8 * h;
  bf16x8 c0 = *(const bf16x8*)p;
  bf16x8 c1 = *(const bf16x8*)(p + 16);
  bf16x16 f;
#pragma unroll
  for (int i = 0; i < 8; ++i) { f[i] = c0[i]; f[8 + i] = c1[i]; }
  return f;
}

// ---------------------------------------------------------------------------
// Generic GEMM: C[M,N] = A[M,K](bf16) * W[N,K]^T(bf16) + bias, optional ELU,
// optional f32 and/or bf16 outputs.
// Register-blocked: each wave owns a 32x64 tile (2 M-subtiles x 4 N-subtiles,
// 8 WMMAs per K-step from 2 A-frags + 4 B-frags). Block = 4 waves in a 2x2
// arrangement -> 64 rows x 128 cols per block.
// Requires M%64==0, N%128==0, K%32==0 (true for every GEMM here).
// ---------------------------------------------------------------------------
__global__ __launch_bounds__(128) void gemm_bf16_kernel(
    const __bf16* __restrict__ A, int lda,
    const __bf16* __restrict__ Bw, int ldb,
    const float* __restrict__ bias,
    float* __restrict__ Cf, __bf16* __restrict__ Cbf, int ldc,
    int M, int N, int K, int flags) {
  const int lane = threadIdx.x & 31;
  const int wave = threadIdx.x >> 5;
  const int row0 = blockIdx.x * 64 + (wave >> 1) * 32;
  const int col0 = blockIdx.y * 128 + (wave & 1) * 64;
  if (row0 >= M || col0 >= N) return;

  const f32x8 zero = {0.f, 0.f, 0.f, 0.f, 0.f, 0.f, 0.f, 0.f};
  f32x8 acc[2][4];
#pragma unroll
  for (int i = 0; i < 2; ++i)
#pragma unroll
    for (int j = 0; j < 4; ++j) acc[i][j] = zero;

  for (int k0 = 0; k0 < K; k0 += 32) {
    bf16x16 af[2];
#pragma unroll
    for (int i = 0; i < 2; ++i) af[i] = frag_ld(A, lda, row0 + 16 * i, k0, lane);
    bf16x16 bf[4];
#pragma unroll
    for (int j = 0; j < 4; ++j) bf[j] = frag_ld(Bw, ldb, col0 + 16 * j, k0, lane);
    if (k0 + 32 < K) {
      __builtin_prefetch(A  + (long)(row0 + (lane & 15)) * lda + k0 + 32, 0, 3);
      __builtin_prefetch(Bw + (long)(col0 + (lane & 15)) * ldb + k0 + 32, 0, 3);
    }
#pragma unroll
    for (int i = 0; i < 2; ++i)
#pragma unroll
      for (int j = 0; j < 4; ++j)
        acc[i][j] = __builtin_amdgcn_wmma_f32_16x16x32_bf16(
            false, af[i], false, bf[j], (short)0, acc[i][j], false, false);
  }

  const int h = lane >> 4, n = lane & 15;
#pragma unroll
  for (int j = 0; j < 4; ++j) {
    float bv = bias ? bias[col0 + 16 * j + n] : 0.f;
#pragma unroll
    for (int i = 0; i < 2; ++i) {
#pragma unroll
      for (int r = 0; r < 8; ++r) {
        int row = row0 + 16 * i + r + 8 * h;
        float v = acc[i][j][r] + bv;
        if (flags & 1) v = (v > 0.f) ? v : (__expf(v) - 1.f);  // ELU
        long idx = (long)row * ldc + col0 + 16 * j + n;
        if (Cf)  Cf[idx]  = v;
        if (Cbf) Cbf[idx] = (__bf16)v;
      }
    }
  }
}

// ---------------------------------------------------------------------------
// Fused graph attention: one wave per (batch, 16-row query tile).
// Q fragments hoisted into registers (loaded once, reused by all 7 key
// tiles). scores(QK^T) -> mask -> row softmax -> probs to LDS (bf16, K
// padded to 128 with zeros) -> msg = P @ V using pre-transposed V.
// Writes msg into cat[:, 512:1024] (bf16).
// ---------------------------------------------------------------------------
__global__ __launch_bounds__(32) void attn_kernel(
    const __bf16* __restrict__ q,   // [B*100, 512]
    const __bf16* __restrict__ k,   // [B*100, 512]
    const __bf16* __restrict__ vT,  // [B, 512, 128] (keys padded w/ zeros)
    const int* __restrict__ entity_num,
    __bf16* __restrict__ catbuf) {  // [B*100, 1024]
  __shared__ __bf16 probs[16][136];
  const int lane = threadIdx.x;
  const int b  = blockIdx.x / 7;
  const int qt = blockIdx.x % 7;
  const int h = lane >> 4, nn = lane & 15;
  const __bf16* qb = q  + (long)b * NNE * CTXE;
  const __bf16* kb = k  + (long)b * NNE * CTXE;
  const __bf16* vb = vT + (long)b * CTXE * 128;
  const int ent = entity_num[b];
  const float scale = 0.044194173824159216f;  // 1/sqrt(512)

  // hoist the 16 Q fragments (K=512) -- reused across all key tiles
  bf16x16 qf[16];
#pragma unroll
  for (int ks = 0; ks < 16; ++ks)
    qf[ks] = frag_ld_clamp(qb, CTXE, qt * 16, ks * 32, lane, NNE);

  float sc[7][8];
#pragma unroll
  for (int kt = 0; kt < 7; ++kt) {
    f32x8 acc = {0.f, 0.f, 0.f, 0.f, 0.f, 0.f, 0.f, 0.f};
#pragma unroll
    for (int ks = 0; ks < 16; ++ks) {
      bf16x16 bf = frag_ld_clamp(kb, CTXE, kt * 16, ks * 32, lane, NNE);
      acc = __builtin_amdgcn_wmma_f32_16x16x32_bf16(false, qf[ks], false, bf,
                                                    (short)0, acc, false, false);
    }
    int col = kt * 16 + nn;
#pragma unroll
    for (int r = 0; r < 8; ++r) {
      int row = qt * 16 + r + 8 * h;
      float v;
      if (col >= NNE)                                  v = -INFINITY; // pad col
      else if (row >= NNE || row >= ent || col >= ent) v = NEGV;      // mask2d
      else                                             v = acc[r] * scale;
      sc[kt][r] = v;
    }
  }
  // row softmax over 7 tiles; rows live in 16-lane halves -> xor-reduce 1..8
#pragma unroll
  for (int r = 0; r < 8; ++r) {
    float mx = -INFINITY;
#pragma unroll
    for (int t = 0; t < 7; ++t) mx = fmaxf(mx, sc[t][r]);
#pragma unroll
    for (int d = 1; d < 16; d <<= 1) mx = fmaxf(mx, __shfl_xor(mx, d, 32));
    float s = 0.f;
#pragma unroll
    for (int t = 0; t < 7; ++t) { float e = __expf(sc[t][r] - mx); sc[t][r] = e; s += e; }
#pragma unroll
    for (int d = 1; d < 16; d <<= 1) s += __shfl_xor(s, d, 32);
    float inv = 1.f / s;
#pragma unroll
    for (int t = 0; t < 7; ++t)
      probs[r + 8 * h][t * 16 + nn] = (__bf16)(sc[t][r] * inv);
  }
  // zero K-padding cols 112..127 (LDS ops are in-order within a wave)
  for (int j = lane; j < 256; j += 32)
    probs[j >> 4][112 + (j & 15)] = (__bf16)0.f;

  // msg = P[16x128] @ V[128x512]; V supplied transposed: vb[c][key]
  for (int ct = 0; ct < 32; ++ct) {
    f32x8 acc = {0.f, 0.f, 0.f, 0.f, 0.f, 0.f, 0.f, 0.f};
#pragma unroll
    for (int ks = 0; ks < 128; ks += 32) {
      bf16x16 a;
      const __bf16* p = &probs[nn][ks + 8 * h];
#pragma unroll
      for (int i = 0; i < 8; ++i) { a[i] = p[i]; a[8 + i] = p[16 + i]; }
      bf16x16 bf = frag_ld(vb, 128, ct * 16, ks, lane);
      acc = __builtin_amdgcn_wmma_f32_16x16x32_bf16(false, a, false, bf,
                                                    (short)0, acc, false, false);
    }
#pragma unroll
    for (int r = 0; r < 8; ++r) {
      int row = qt * 16 + r + 8 * h;
      if (row < NNE)
        catbuf[(long)(b * NNE + row) * 1024 + 512 + ct * 16 + nn] = (__bf16)acc[r];
    }
  }
}

// ---------------------------------------------------------------------------
// Elementwise / small kernels
// ---------------------------------------------------------------------------
__global__ void cvt_f32_bf16(const float* __restrict__ s, __bf16* __restrict__ d, long n) {
  long i = (long)blockIdx.x * blockDim.x + threadIdx.x;
  if (i < n) d[i] = (__bf16)s[i];
}

__global__ __launch_bounds__(256) void normalize_kernel(
    const float* __restrict__ imgs, __bf16* __restrict__ out) {
  long row = blockIdx.x;
  const float* src = imgs + row * DFE;
  __shared__ float red[8];
  float s = 0.f;
  for (int i = threadIdx.x; i < DFE; i += 256) { float v = src[i]; s += v * v; }
#pragma unroll
  for (int d = 1; d < 32; d <<= 1) s += __shfl_xor(s, d, 32);
  if ((threadIdx.x & 31) == 0) red[threadIdx.x >> 5] = s;
  __syncthreads();
  float tot = 0.f;
#pragma unroll
  for (int w = 0; w < 8; ++w) tot += red[w];
  float inv = 1.f / fmaxf(sqrtf(tot), 1e-12f);
  __bf16* dst = out + row * DFE;
  for (int i = threadIdx.x; i < DFE; i += 256) dst[i] = (__bf16)(src[i] * inv);
}

__global__ void init_xctx(const float* __restrict__ initMem,
                          float* __restrict__ xf, __bf16* __restrict__ xb) {
  long i = (long)blockIdx.x * blockDim.x + threadIdx.x;
  if (i >= (long)ROWS * CTXE) return;
  float v = initMem[i & (CTXE - 1)];
  xf[i] = v; xb[i] = (__bf16)v;
}

__global__ void raw_kernel(const float* __restrict__ qcmd, const float* __restrict__ lstm,
                           const float* __restrict__ c2lw, const float* __restrict__ c2lb,
                           const int* __restrict__ qlen, float* __restrict__ raw) {
  int gw = (int)(((long)blockIdx.x * blockDim.x + threadIdx.x) >> 5);
  int lane = threadIdx.x & 31;
  if (gw >= BB * LLE) return;
  int b = gw / LLE, l = gw % LLE;
  const float* qc = qcmd + (long)b * CMDE;
  const float* ls = lstm + ((long)b * LLE + l) * CMDE;
  float s = 0.f;
  for (int c = lane; c < CMDE; c += 32) s += qc[c] * ls[c] * c2lw[c];
#pragma unroll
  for (int d = 1; d < 32; d <<= 1) s += __shfl_xor(s, d, 32);
  if (lane == 0) {
    float v = s + c2lb[0];
    if (l >= qlen[b]) v = NEGV;
    raw[b * LLE + l] = v;
  }
}

__global__ __launch_bounds__(128) void cmd_kernel(
    const float* __restrict__ raw, const float* __restrict__ lstm,
    float* __restrict__ cmdf, __bf16* __restrict__ cmdb) {
  int b = blockIdx.x;
  __shared__ float att[LLE];
  if (threadIdx.x < 32) {
    int lane = threadIdx.x;
    float v = (lane < LLE) ? raw[b * LLE + lane] : -INFINITY;
    float mx = v;
#pragma unroll
    for (int d = 1; d < 32; d <<= 1) mx = fmaxf(mx, __shfl_xor(mx, d, 32));
    float e = __expf(v - mx);
    float s = e;
#pragma unroll
    for (int d = 1; d < 32; d <<= 1) s += __shfl_xor(s, d, 32);
    if (lane < LLE) att[lane] = e / s;
  }
  __syncthreads();
  for (int c = threadIdx.x; c < CMDE; c += 128) {
    float s = 0.f;
    for (int l = 0; l < LLE; ++l) s += att[l] * lstm[((long)b * LLE + l) * CMDE + c];
    cmdf[(long)b * CMDE + c] = s;
    cmdb[(long)b * CMDE + c] = (__bf16)s;
  }
}

__global__ void build_xj(const __bf16* __restrict__ xl, const __bf16* __restrict__ xc,
                         const float* __restrict__ pxl, const float* __restrict__ pxc,
                         __bf16* __restrict__ xj) {
  long i = (long)blockIdx.x * blockDim.x + threadIdx.x;
  if (i >= (long)ROWS * CTXE) return;
  long row = i >> 9; int c = (int)(i & 511);
  __bf16* d = xj + row * (3 * CTXE);
  d[c] = xl[i];
  d[CTXE + c] = xc[i];
  d[2 * CTXE + c] = (__bf16)(pxl[i] * pxc[i]);
}

__global__ void modulate_kv(const float* __restrict__ kf, const float* __restrict__ vf,
                            const float* __restrict__ pkf, const float* __restrict__ pvf,
                            __bf16* __restrict__ kb, __bf16* __restrict__ vT) {
  long i = (long)blockIdx.x * blockDim.x + threadIdx.x;
  if (i >= (long)ROWS * CTXE) return;
  long bn = i >> 9; int c = (int)(i & 511);
  int b = (int)(bn / NNE), n = (int)(bn % NNE);
  kb[i] = (__bf16)(kf[i] * pkf[(long)b * CTXE + c]);
  vT[((long)b * CTXE + c) * 128 + n] = (__bf16)(vf[i] * pvf[(long)b * CTXE + c]);
}

__global__ void zero_vT_pad(__bf16* __restrict__ vT) {
  long i = (long)blockIdx.x * blockDim.x + threadIdx.x;
  long total = (long)BB * CTXE * 28;
  if (i >= total) return;
  long b = i / (CTXE * 28); long rem = i % (CTXE * 28);
  long c = rem / 28; int j = (int)(rem % 28);
  vT[((long)b * CTXE + c) * 128 + NNE + j] = (__bf16)0.f;
}

__global__ void copy_into_cat(const __bf16* __restrict__ src, __bf16* __restrict__ cat, int off) {
  long i = (long)blockIdx.x * blockDim.x + threadIdx.x;
  if (i >= (long)ROWS * CTXE) return;
  long row = i >> 9; int c = (int)(i & 511);
  cat[row * 1024 + off + c] = src[i];
}

// ---------------------------------------------------------------------------
// Host side
// ---------------------------------------------------------------------------
static inline char* bump(char*& p, size_t bytes) {
  char* r = p; p += (bytes + 255) & ~(size_t)255; return r;
}
static void launch_gemm(hipStream_t s, const __bf16* A, int lda, const __bf16* Bw, int ldb,
                        const float* bias, float* Cf, __bf16* Cbf, int ldc,
                        int M, int N, int K, int flags) {
  dim3 g(M / 64, N / 128), b(128);
  hipLaunchKernelGGL(gemm_bf16_kernel, g, b, 0, s, A, lda, Bw, ldb, bias, Cf, Cbf, ldc, M, N, K, flags);
}
static void launch_cvt(hipStream_t s, const float* src, __bf16* dst, long n) {
  hipLaunchKernelGGL(cvt_f32_bf16, dim3((n + 255) / 256), dim3(256), 0, s, src, dst, n);
}

extern "C" void kernel_launch(void* const* d_in, const int* in_sizes, int n_in,
                              void* d_out, int out_size, void* d_ws, size_t ws_size,
                              hipStream_t stream) {
  (void)in_sizes; (void)n_in; (void)out_size; (void)ws_size;
  const float* images     = (const float*)d_in[0];
  const float* q_encoding = (const float*)d_in[1];
  const float* lstm       = (const float*)d_in[2];
  const int*   q_length   = (const int*)d_in[3];
  const int*   entity_num = (const int*)d_in[4];
  const float* initKB_W = (const float*)d_in[5];  const float* initKB_b = (const float*)d_in[6];
  const float* initMem  = (const float*)d_in[7];
  const float* qInput_W = (const float*)d_in[8];  const float* qInput_b = (const float*)d_in[9];
  const float* qInput2_W = (const float*)d_in[10]; const float* qInput2_b = (const float*)d_in[11];
  const float* c2l_W = (const float*)d_in[12];    const float* c2l_b = (const float*)d_in[13];
  const float* pxl_W = (const float*)d_in[14];    const float* pxl_b = (const float*)d_in[15];
  const float* pxc_W = (const float*)d_in[16];    const float* pxc_b = (const float*)d_in[17];
  const float* qry_W = (const float*)d_in[18];    const float* qry_b = (const float*)d_in[19];
  const float* key_W = (const float*)d_in[20];    const float* key_b = (const float*)d_in[21];
  const float* val_W = (const float*)d_in[22];    const float* val_b = (const float*)d_in[23];
  const float* pk_W  = (const float*)d_in[24];    const float* pk_b  = (const float*)d_in[25];
  const float* pv_W  = (const float*)d_in[26];    const float* pv_b  = (const float*)d_in[27];
  const float* mu_W  = (const float*)d_in[28];    const float* mu_b  = (const float*)d_in[29];
  const float* ckb_W = (const float*)d_in[30];    const float* ckb_b = (const float*)d_in[31];

  char* p = (char*)d_ws;
  __bf16* imgsn = (__bf16*)bump(p, (long)ROWS * DFE * 2);
  __bf16* Wkb   = (__bf16*)bump(p, (long)CTXE * DFE * 2);
  __bf16* Wqin  = (__bf16*)bump(p, (long)CMDE * CMDE * 2);
  __bf16* Wqin2 = (__bf16*)bump(p, (long)TTE * CMDE * CMDE * 2);
  __bf16* Wpxl  = (__bf16*)bump(p, (long)CTXE * CTXE * 2);
  __bf16* Wpxc  = (__bf16*)bump(p, (long)CTXE * CTXE * 2);
  __bf16* Wqry  = (__bf16*)bump(p, (long)CTXE * 3 * CTXE * 2);
  __bf16* Wkey  = (__bf16*)bump(p, (long)CTXE * 3 * CTXE * 2);
  __bf16* Wval  = (__bf16*)bump(p, (long)CTXE * 3 * CTXE * 2);
  __bf16* Wpk   = (__bf16*)bump(p, (long)CTXE * CMDE * 2);
  __bf16* Wpv   = (__bf16*)bump(p, (long)CTXE * CMDE * 2);
  __bf16* Wmu   = (__bf16*)bump(p, (long)CTXE * 2 * CTXE * 2);
  __bf16* Wckb  = (__bf16*)bump(p, (long)CTXE * 2 * CTXE * 2);
  float*  xlocf = (float*)bump(p, (long)ROWS * CTXE * 4);
  __bf16* xlocb = (__bf16*)bump(p, (long)ROWS * CTXE * 2);
  float*  xctxf = (float*)bump(p, (long)ROWS * CTXE * 4);
  __bf16* xctxb = (__bf16*)bump(p, (long)ROWS * CTXE * 2);
  __bf16* qencb = (__bf16*)bump(p, (long)BB * CMDE * 2);
  float*  qpref = (float*)bump(p, (long)BB * CMDE * 4);
  __bf16* qpreb = (__bf16*)bump(p, (long)BB * CMDE * 2);
  float*  qcmdf = (float*)bump(p, (long)BB * CMDE * 4);
  float*  rawb  = (float*)bump(p, (long)BB * LLE * 4);
  float*  cmdf  = (float*)bump(p, (long)BB * CMDE * 4);
  __bf16* cmdb  = (__bf16*)bump(p, (long)BB * CMDE * 2);
  float*  pkfac = (float*)bump(p, (long)BB * CTXE * 4);
  float*  pvfac = (float*)bump(p, (long)BB * CTXE * 4);
  float*  pxlf  = (float*)bump(p, (long)ROWS * CTXE * 4);
  float*  pxcf  = (float*)bump(p, (long)ROWS * CTXE * 4);
  __bf16* xj    = (__bf16*)bump(p, (long)ROWS * 3 * CTXE * 2);
  __bf16* qbf   = (__bf16*)bump(p, (long)ROWS * CTXE * 2);
  float*  kf    = (float*)bump(p, (long)ROWS * CTXE * 4);
  float*  vf    = (float*)bump(p, (long)ROWS * CTXE * 4);
  __bf16* kbf   = (__bf16*)bump(p, (long)ROWS * CTXE * 2);
  __bf16* vT    = (__bf16*)bump(p, (long)BB * CTXE * 128 * 2);
  __bf16* catb  = (__bf16*)bump(p, (long)ROWS * 2 * CTXE * 2);

  const long RC = (long)ROWS * CTXE;
  const int EB = 256;

  // weights -> bf16
  launch_cvt(stream, initKB_W, Wkb, (long)CTXE * DFE);
  launch_cvt(stream, qInput_W, Wqin, (long)CMDE * CMDE);
  launch_cvt(stream, qInput2_W, Wqin2, (long)TTE * CMDE * CMDE);
  launch_cvt(stream, pxl_W, Wpxl, (long)CTXE * CTXE);
  launch_cvt(stream, pxc_W, Wpxc, (long)CTXE * CTXE);
  launch_cvt(stream, qry_W, Wqry, (long)CTXE * 3 * CTXE);
  launch_cvt(stream, key_W, Wkey, (long)CTXE * 3 * CTXE);
  launch_cvt(stream, val_W, Wval, (long)CTXE * 3 * CTXE);
  launch_cvt(stream, pk_W, Wpk, (long)CTXE * CMDE);
  launch_cvt(stream, pv_W, Wpv, (long)CTXE * CMDE);
  launch_cvt(stream, mu_W, Wmu, (long)CTXE * 2 * CTXE);
  launch_cvt(stream, ckb_W, Wckb, (long)CTXE * 2 * CTXE);
  launch_cvt(stream, q_encoding, qencb, (long)BB * CMDE);

  // stem: normalize + project
  hipLaunchKernelGGL(normalize_kernel, dim3(ROWS), dim3(256), 0, stream, images, imgsn);
  launch_gemm(stream, imgsn, DFE, Wkb, DFE, initKB_b, xlocf, xlocb, CTXE, ROWS, CTXE, DFE, 0);
  hipLaunchKernelGGL(init_xctx, dim3((RC + EB - 1) / EB), dim3(EB), 0, stream, initMem, xctxf, xctxb);

  // q_pre = elu(q_enc @ Wqin^T + b)
  launch_gemm(stream, qencb, CMDE, Wqin, CMDE, qInput_b, qpref, qpreb, CMDE, BB, CMDE, CMDE, 1);

  for (int t = 0; t < TTE; ++t) {
    launch_gemm(stream, qpreb, CMDE, Wqin2 + (long)t * CMDE * CMDE, CMDE,
                qInput2_b + (long)t * CMDE, qcmdf, nullptr, CMDE, BB, CMDE, CMDE, 0);
    hipLaunchKernelGGL(raw_kernel, dim3((BB * LLE * 32 + 255) / 256), dim3(256), 0, stream,
                       qcmdf, lstm, c2l_W, c2l_b, q_length, rawb);
    hipLaunchKernelGGL(cmd_kernel, dim3(BB), dim3(128), 0, stream, rawb, lstm, cmdf, cmdb);
    launch_gemm(stream, cmdb, CMDE, Wpk, CMDE, pk_b, pkfac, nullptr, CTXE, BB, CTXE, CMDE, 0);
    launch_gemm(stream, cmdb, CMDE, Wpv, CMDE, pv_b, pvfac, nullptr, CTXE, BB, CTXE, CMDE, 0);

    launch_gemm(stream, xlocb, CTXE, Wpxl, CTXE, pxl_b, pxlf, nullptr, CTXE, ROWS, CTXE, CTXE, 0);
    launch_gemm(stream, xctxb, CTXE, Wpxc, CTXE, pxc_b, pxcf, nullptr, CTXE, ROWS, CTXE, CTXE, 0);
    hipLaunchKernelGGL(build_xj, dim3((RC + EB - 1) / EB), dim3(EB), 0, stream,
                       xlocb, xctxb, pxlf, pxcf, xj);

    launch_gemm(stream, xj, 3 * CTXE, Wqry, 3 * CTXE, qry_b, nullptr, qbf, CTXE, ROWS, CTXE, 3 * CTXE, 0);
    launch_gemm(stream, xj, 3 * CTXE, Wkey, 3 * CTXE, key_b, kf, nullptr, CTXE, ROWS, CTXE, 3 * CTXE, 0);
    launch_gemm(stream, xj, 3 * CTXE, Wval, 3 * CTXE, val_b, vf, nullptr, CTXE, ROWS, CTXE, 3 * CTXE, 0);
    hipLaunchKernelGGL(modulate_kv, dim3((RC + EB - 1) / EB), dim3(EB), 0, stream,
                       kf, vf, pkfac, pvfac, kbf, vT);
    hipLaunchKernelGGL(zero_vT_pad, dim3(((long)BB * CTXE * 28 + EB - 1) / EB), dim3(EB), 0, stream, vT);

    hipLaunchKernelGGL(copy_into_cat, dim3((RC + EB - 1) / EB), dim3(EB), 0, stream, xctxb, catb, 0);
    hipLaunchKernelGGL(attn_kernel, dim3(BB * 7), dim3(32), 0, stream, qbf, kbf, vT, entity_num, catb);

    launch_gemm(stream, catb, 2 * CTXE, Wmu, 2 * CTXE, mu_b, xctxf, xctxb, CTXE, ROWS, CTXE, 2 * CTXE, 0);
  }

  // final: concat(x_loc, x_ctx) @ ckb^T + b -> d_out (f32)
  hipLaunchKernelGGL(copy_into_cat, dim3((RC + EB - 1) / EB), dim3(EB), 0, stream, xlocb, catb, 0);
  hipLaunchKernelGGL(copy_into_cat, dim3((RC + EB - 1) / EB), dim3(EB), 0, stream, xctxb, catb, CTXE);
  launch_gemm(stream, catb, 2 * CTXE, Wckb, 2 * CTXE, ckb_b, (float*)d_out, nullptr, CTXE,
              ROWS, CTXE, 2 * CTXE, 0);
}
